// EnhancedQCNN_78821239816247
// MI455X (gfx1250) — matrix operand order repaired and checked by reading, not compiled, not run on hardware
//
#include <hip/hip_runtime.h>
#include <hip/hip_bf16.h>

// ---------------------------------------------------------------------------
// EnhancedQCNN forward for MI455X (gfx1250, wave32, WMMA).
//
// Pipeline (all activations f16, f32 accumulate in WMMA):
//   conv1a(1->16) -> conv1b(16->16) -> pool -> conv2a(16->32) -> conv2b(32->32)
//   -> pool -> quantum branch (16x16 complex unitary per 2x2 patch + pooling)
//   -> MHA (WMMA QK^T / PV) -> mean -> fc1(+BN) -> fc2 -> fc3 -> d_out (16x10 f32)
//
// Convs = implicit GEMM: 1 wave computes a 16(pixel)xCOUT(chan) tile (COUT/16 WMMA tiles),
// K = Cin*9 in 32-wide WMMA steps; input halo staged in LDS via
// global_load_async_to_lds_b128 (ASYNCcnt) when available; weights
// pre-swizzled into CDNA5 B-fragment order (one 32B load per k-step).
// BN+bias+ReLU folded into per-channel scale/shift epilogues.
// ---------------------------------------------------------------------------

typedef __attribute__((ext_vector_type(16))) _Float16 v16h;
typedef __attribute__((ext_vector_type(8)))  _Float16 v8h;
typedef __attribute__((ext_vector_type(8)))  float    v8f;

#define V8F_ZERO (v8f){0.f,0.f,0.f,0.f,0.f,0.f,0.f,0.f}

#if defined(__AMDGCN__) && \
    __has_builtin(__builtin_amdgcn_global_load_async_to_lds_b128) && \
    __has_builtin(__builtin_amdgcn_s_wait_asynccnt)
#define USE_ASYNC_LDS 1
#else
#define USE_ASYNC_LDS 0
#endif

#if USE_ASYNC_LDS
// Builtin signature (from hipcc diagnostic): param0 = AS1 pointer to
// 'int __attribute__((vector_size(16)))' (printed as '__device__ int4*').
typedef int v4i_vs __attribute__((vector_size(4 * sizeof(int))));
typedef __attribute__((address_space(1))) v4i_vs* as1_v4p;
typedef __attribute__((address_space(3))) v4i_vs* as3_v4p;

__device__ __forceinline__ void async_g2l_b128(const _Float16* g, _Float16* l) {
  __builtin_amdgcn_global_load_async_to_lds_b128(
      (as1_v4p)(unsigned long long)g,
      (as3_v4p)(unsigned)(unsigned long long)l,
      0, 0);
}
#endif

// ---------------- weight -> B-fragment swizzle (W is (N,K) row-major f32) ----
__global__ void k_prep_wfrag(const float* __restrict__ w, _Float16* __restrict__ frag,
                             int N, int K) {
  int KB = (K + 31) >> 5;
  int NT = (N + 15) >> 4;
  int tot = NT * KB * 32;
  int idx = blockIdx.x * 256 + threadIdx.x;
  if (idx >= tot) return;
  int lane = idx & 31;
  int kb   = (idx >> 5) % KB;
  int t    = idx / (32 * KB);
  int n    = t * 16 + (lane & 15);
  int hi   = lane >> 4;
  #pragma unroll
  for (int i = 0; i < 16; ++i) {
    int kk = kb * 32 + ((i < 8) ? (8 * hi + i) : (16 + 8 * hi + (i - 8)));   // B layout: half 0-7 -> K hi*8+i, 8-15 -> K 16+hi*8+i-8
    float v = (n < N && kk < K) ? w[(size_t)n * K + kk] : 0.f;
    frag[(size_t)idx * 16 + i] = (_Float16)v;
  }
}

// ---------------- fold conv-bias + BN into scale/shift -----------------------
__global__ void k_bnprep(const float* g, const float* bb, const float* m,
                         const float* v, const float* cb,
                         float* scale, float* shift, int n) {
  int i = blockIdx.x * 256 + threadIdx.x;
  if (i >= n) return;
  float s = g[i] * rsqrtf(v[i] + 1e-5f);
  scale[i] = s;
  shift[i] = ((cb ? cb[i] : 0.f) - m[i]) * s + bb[i];
}

// ---------------- implicit-GEMM 3x3 conv + BN + ReLU (WMMA) ------------------
// LDS tile: per (c,r) halo row, 32-half stride; x in [0,18) stored at 7+x so
// the always-in-bounds interior (x=1..16 i.e. gx=w0..w0+15) is 16B aligned.
template <int CIN, int COUT, bool INF32>
__global__ void __launch_bounds__(32)
k_conv3x3_wmma(const void* __restrict__ inp, const _Float16* __restrict__ wfrag,
               const float* __restrict__ scale, const float* __restrict__ shift,
               _Float16* __restrict__ out, int B, int H, int W) {
  constexpr int KTOT = CIN * 9;
  constexpr int KB   = (KTOT + 31) / 32;
  constexpr int NT   = COUT / 16;
  constexpr int RS   = 32;                       // row stride (halves)
  __shared__ __attribute__((aligned(64))) _Float16 tile[CIN * 3 * RS];

  int lane = threadIdx.x;
  int wt = W >> 4;
  int gid = blockIdx.x;
  int w0 = (gid % wt) << 4;
  int h  = (gid / wt) % H;
  int b  = gid / (wt * H);

  if constexpr (!INF32) {
    const _Float16* inh = (const _Float16*)inp;
    for (int rid = lane; rid < CIN * 3; rid += 32) {
      int c = rid / 3, r = rid % 3;
      int y = h + r - 1;
      _Float16* lrow = &tile[rid * RS];
      if (y >= 0 && y < H) {
        const _Float16* grow = inh + ((size_t)(b * CIN + c) * H + y) * W + w0;
#if USE_ASYNC_LDS
        async_g2l_b128(grow,     lrow + 8);      // gx = w0..w0+7
        async_g2l_b128(grow + 8, lrow + 16);     // gx = w0+8..w0+15
#else
        *(v8h*)(lrow + 8)  = *(const v8h*)grow;
        *(v8h*)(lrow + 16) = *(const v8h*)(grow + 8);
#endif
        lrow[7]  = (w0 > 0)      ? grow[-1] : (_Float16)0.f;   // gx = w0-1
        lrow[24] = (w0 + 16 < W) ? grow[16] : (_Float16)0.f;   // gx = w0+16
      } else {
        #pragma unroll
        for (int xx = 0; xx < 18; ++xx) lrow[7 + xx] = (_Float16)0.f;
      }
    }
  } else {
    const float* inf = (const float*)inp;
    for (int idx = lane; idx < CIN * 54; idx += 32) {
      int c = idx / 54, rem = idx % 54, r = rem / 18, xx = rem % 18;
      int y = h + r - 1, gx = w0 + xx - 1;
      float v = 0.f;
      if (y >= 0 && y < H && gx >= 0 && gx < W)
        v = inf[((size_t)(b * CIN + c) * H + y) * W + gx];
      tile[(c * 3 + r) * RS + 7 + xx] = (_Float16)v;
    }
  }
#if USE_ASYNC_LDS
  __builtin_amdgcn_s_wait_asynccnt(0);
#endif
  __syncthreads();

  int m = lane & 15, hi = lane >> 4;
  v8f acc[NT];
  #pragma unroll
  for (int t = 0; t < NT; ++t) acc[t] = V8F_ZERO;

  const v16h* wf = (const v16h*)wfrag;
  for (int kb = 0; kb < KB; ++kb) {
    if (kb + 1 < KB)
      __builtin_prefetch(wf + (size_t)(kb + 1) * 32 + lane, 0, 0);
    v16h a;
    #pragma unroll
    for (int i = 0; i < 16; ++i) {     // A layout: half 0-7 -> K hi*8+i, 8-15 -> K 16+hi*8+i-8
      int kk = kb * 32 + ((i < 8) ? (hi * 8 + i) : (16 + hi * 8 + (i - 8)));
      _Float16 v = (_Float16)0.f;
      if (kk < KTOT) {
        int c = kk / 9, rs = kk % 9, r = rs / 3, s = rs % 3;
        v = tile[(c * 3 + r) * RS + 7 + m + s];
      }
      a[i] = v;
    }
    #pragma unroll
    for (int t = 0; t < NT; ++t) {
      v16h bf = wf[(t * KB + kb) * 32 + lane];
      acc[t] = __builtin_amdgcn_wmma_f32_16x16x32_f16(false, a, false, bf,
                                                      (short)0, acc[t], false, false);
    }
  }

  #pragma unroll
  for (int t = 0; t < NT; ++t) {
    int ch = t * 16 + m;               // D layout: col = lane&15
    float sc = scale[ch], sh = shift[ch];
    #pragma unroll
    for (int g = 0; g < 8; ++g) {
      float val = fmaxf(acc[t][g] * sc + sh, 0.f);
      int mm = g + 8 * hi;             // D layout: row = g + 8*(lane>>4)
      out[((size_t)(b * COUT + ch) * H + h) * W + w0 + mm] = (_Float16)val;
    }
  }
}

// ---------------- 2x2 maxpool (f16) -----------------------------------------
__global__ void k_pool2(const _Float16* __restrict__ in, _Float16* __restrict__ out,
                        int BC, int Ho, int Wo) {
  int idx = blockIdx.x * 256 + threadIdx.x;
  if (idx >= BC * Ho * Wo) return;
  int wo = idx % Wo, ho = (idx / Wo) % Ho, bc = idx / (Wo * Ho);
  size_t base = ((size_t)bc * (2 * Ho) + 2 * ho) * (2 * Wo) + 2 * wo;
  float a = (float)in[base],        b = (float)in[base + 1];
  float c = (float)in[base + 2*Wo], d = (float)in[base + 2*Wo + 1];
  out[idx] = (_Float16)fmaxf(fmaxf(a, b), fmaxf(c, d));
}

// ---------------- generic WMMA GEMM: out = act(A @ W^T + b) [* s + t] --------
// A: (M,K) f16 row-major, K % 32 == 0, M % 16 == 0. grid = (M/16, ceil(N/16)).
template <int ACT>  // 0 = none, 1 = relu
__global__ void __launch_bounds__(32)
k_gemm16(const _Float16* __restrict__ A, int K, const _Float16* __restrict__ wfrag,
         const float* __restrict__ bias, const float* __restrict__ scale,
         const float* __restrict__ shift, _Float16* __restrict__ outh,
         float* __restrict__ outf, int ldo, int Nvalid) {
  int lane = threadIdx.x, m = lane & 15, hi = lane >> 4;
  int m0 = blockIdx.x * 16, t = blockIdx.y;
  int KB = K >> 5;
  v8f acc = V8F_ZERO;
  const v16h* wf = (const v16h*)wfrag;
  const _Float16* arow = A + (size_t)(m0 + m) * K;
  for (int kb = 0; kb < KB; ++kb) {
    v8h lo = *(const v8h*)(arow + kb * 32 + hi * 8);
    v8h hh = *(const v8h*)(arow + kb * 32 + 16 + hi * 8);
    v16h a;
    #pragma unroll
    for (int i = 0; i < 8; ++i) { a[i] = lo[i]; a[i + 8] = hh[i]; }
    v16h bf = wf[(t * KB + kb) * 32 + lane];
    acc = __builtin_amdgcn_wmma_f32_16x16x32_f16(false, a, false, bf,
                                                 (short)0, acc, false, false);
  }
  int n = t * 16 + m;
  if (n < Nvalid) {
    float bsum = bias[n];
    float sc = scale ? scale[n] : 1.f;
    float sh = shift ? shift[n] : 0.f;
    #pragma unroll
    for (int g = 0; g < 8; ++g) {
      float val = acc[g] + bsum;
      if (ACT == 1) val = fmaxf(val, 0.f);
      if (scale) val = val * sc + sh;
      int r = m0 + g + 8 * hi;
      if (outh) outh[(size_t)r * ldo + n] = (_Float16)val;
      if (outf) outf[(size_t)r * ldo + n] = val;
    }
  }
}

// ---------------- quantum branch ---------------------------------------------
// Build 16x16 complex unitary from qparams (one wave, thread j owns column j).
__device__ __forceinline__ int ring_src(int i) {
  // RING = C(3,0) C(2,3) C(1,2) C(0,1); (RING U)[i] = U[a01(a12(a23(a30(i))))]
  int s = i;
  s = ((s >> 0) & 1) ? (s ^ (1 << 3)) : s;  // a30: ctrl bit0, tgt bit3
  s = ((s >> 1) & 1) ? (s ^ (1 << 0)) : s;  // a23
  s = ((s >> 2) & 1) ? (s ^ (1 << 1)) : s;  // a12
  s = ((s >> 3) & 1) ? (s ^ (1 << 2)) : s;  // a01
  return s;
}

__global__ void k_qunitary(const float* __restrict__ qp,
                           float* __restrict__ Ur, float* __restrict__ Ui) {
  int j = threadIdx.x;
  if (j >= 16) return;
  float ur[16], ui[16], nr[16], ni[16];
  for (int i = 0; i < 16; ++i) { ur[i] = (i == j) ? 1.f : 0.f; ui[i] = 0.f; }
  for (int layer = 0; layer < 2; ++layer) {
    for (int q = 0; q < 4; ++q) {
      const float* p = qp + (layer * 4 + q) * 3;
      float phi = p[0], th = p[1], om = p[2];
      float ch = cosf(0.5f * th), sh = sinf(0.5f * th);
      float apo = 0.5f * (phi + om), amo = 0.5f * (phi - om);
      float epr = cosf(apo), epi = -sinf(apo);
      float emr = cosf(amo), emi = -sinf(amo);
      float g00r =  epr * ch, g00i =  epi * ch;
      float g01r = -emr * sh, g01i =  emi * sh;
      float g10r =  emr * sh, g10i =  emi * sh;
      float g11r =  epr * ch, g11i = -epi * ch;
      int bp = 3 - q;
      for (int i = 0; i < 16; ++i) {
        int i0 = i & ~(1 << bp), i1 = i | (1 << bp);
        int bs = (i >> bp) & 1;
        float ar = bs ? g10r : g00r, ai = bs ? g10i : g00i;
        float br = bs ? g11r : g01r, bi = bs ? g11i : g01i;
        nr[i] = ar * ur[i0] - ai * ui[i0] + br * ur[i1] - bi * ui[i1];
        ni[i] = ar * ui[i0] + ai * ur[i0] + br * ui[i1] + bi * ur[i1];
      }
      for (int i = 0; i < 16; ++i) { ur[i] = nr[i]; ui[i] = ni[i]; }
    }
    for (int i = 0; i < 16; ++i) { int s = ring_src(i); nr[i] = ur[s]; ni[i] = ui[s]; }
    for (int i = 0; i < 16; ++i) { ur[i] = nr[i]; ui[i] = ni[i]; }
  }
  for (int i = 0; i < 16; ++i) { Ur[i * 16 + j] = ur[i]; Ui[i * 16 + j] = ui[i]; }
}

// Stage 1: per 2x2 patch -> |U st|^2 @ SIGNS -> qc_t (b,c,f,32,32) f32
__global__ void k_qstage1(const _Float16* __restrict__ x2, const float* __restrict__ Ur,
                          const float* __restrict__ Ui, float* __restrict__ qct) {
  int t = blockIdx.x * 256 + threadIdx.x;
  if (t >= 16 * 8 * 32 * 32) return;
  int w = t & 31, h = (t >> 5) & 31, c = (t >> 10) & 7, b = t >> 13;
  size_t base = ((size_t)(b * 32 + c) * 64 + 2 * h) * 64 + 2 * w;
  float v0 = (float)x2[base],      v1 = (float)x2[base + 1];
  float v2 = (float)x2[base + 64], v3 = (float)x2[base + 65];
  float inv = 1.f / (sqrtf(v0*v0 + v1*v1 + v2*v2 + v3*v3) + 1e-8f);
  float s0 = v0*inv, s1 = v1*inv, s2 = v2*inv, s3 = v3*inv;
  float qc0 = 0.f, qc1 = 0.f, qc2 = 0.f, qc3 = 0.f;
  #pragma unroll
  for (int f = 0; f < 16; ++f) {
    const float* rr = Ur + f * 16;
    const float* ii = Ui + f * 16;
    float ar = rr[0]*s0 + rr[1]*s1 + rr[2]*s2 + rr[3]*s3;
    float ai = ii[0]*s0 + ii[1]*s1 + ii[2]*s2 + ii[3]*s3;
    float p = ar*ar + ai*ai;
    qc0 += ((f >> 3) & 1) ? -p : p;  // SIGNS[f][w] = 1-2*((f>>(3-w))&1)
    qc1 += ((f >> 2) & 1) ? -p : p;
    qc2 += ((f >> 1) & 1) ? -p : p;
    qc3 += ((f >> 0) & 1) ? -p : p;
  }
  size_t o = (((size_t)(b * 8 + c) * 4) * 32 + h) * 32 + w;
  qct[o]          = qc0;
  qct[o + 1024]   = qc1;
  qct[o + 2048]   = qc2;
  qct[o + 3072]   = qc3;
}

// Stage 2: second 2x2 pool with UPOOL rotation -> f (16, 8192) f16
__global__ void k_qstage2(const float* __restrict__ qct, _Float16* __restrict__ f) {
  int t = blockIdx.x * 256 + threadIdx.x;
  if (t >= 16 * 8 * 4 * 16 * 16) return;
  int f4 = t & 3, c = (t >> 2) & 7, wp = (t >> 5) & 15, hp = (t >> 9) & 15, b = t >> 13;
  const float* src = qct + ((size_t)(b * 8 + c) * 4 + f4) * 1024;
  float v0 = src[(2*hp) * 32 + 2*wp],     v1 = src[(2*hp) * 32 + 2*wp + 1];
  float v2 = src[(2*hp + 1) * 32 + 2*wp], v3 = src[(2*hp + 1) * 32 + 2*wp + 1];
  float inv = 1.f / (sqrtf(v0*v0 + v1*v1 + v2*v2 + v3*v3) + 1e-8f);
  v0 *= inv; v1 *= inv; v2 *= inv; v3 *= inv;
  const float CR = 0.6532814824381883f;   // cos(pi/8)/sqrt(2)
  const float SR = 0.2705980500730985f;   // sin(pi/8)/sqrt(2)
  float t0 =  CR*v0 - SR*v1 + CR*v2 + SR*v3;
  float t1 = -SR*v0 + CR*v1 + SR*v2 + CR*v3;
  float t2 =  SR*v0 + CR*v1 + SR*v2 - CR*v3;
  float t3 =  CR*v0 + SR*v1 - CR*v2 + SR*v3;
  float z = t0*t0 + t1*t1 - t2*t2 - t3*t3;
  f[(size_t)b * 8192 + hp * 512 + wp * 32 + c * 4 + f4] = (_Float16)z;
}

// ---------------- attention core: one wave per (b, head, 16-query block) -----
__global__ void __launch_bounds__(32)
k_attn(const _Float16* __restrict__ qkv, _Float16* __restrict__ att) {
  __shared__ __attribute__((aligned(16))) _Float16 lp[16 * 64];  // P block
  int lane = threadIdx.x, nloc = lane & 15, hi = lane >> 4;
  int blk = blockIdx.x;
  int b = blk >> 5, h = (blk >> 2) & 7, mblk = blk & 3;
  int s0 = mblk * 16;

  // Q fragment: rows = queries, K-dim = d(16) zero-padded to 32
  const _Float16* qbase = qkv + (size_t)(b * 64 + s0 + nloc) * 384 + h * 16;
  v8h ql = *(const v8h*)(qbase + hi * 8);
  v16h aq;
  #pragma unroll
  for (int i = 0; i < 8; ++i) { aq[i] = ql[i]; aq[i + 8] = (_Float16)0.f; }

  v8f sc[4];
  #pragma unroll
  for (int t = 0; t < 4; ++t) {
    v16h bk;
    const _Float16* kbase = qkv + (size_t)(b * 64 + t * 16 + nloc) * 384 + 128 + h * 16;
    {
      v8h k0 = *(const v8h*)(kbase + hi * 8);
      #pragma unroll
      for (int i = 0; i < 8; ++i) { bk[i] = k0[i]; bk[i + 8] = (_Float16)0.f; }
    }
    v8f z = V8F_ZERO;
    sc[t] = __builtin_amdgcn_wmma_f32_16x16x32_f16(false, aq, false, bk,
                                                   (short)0, z, false, false);
  }

  // scaled softmax per row (row = g + 8*hi; xor masks <=8 stay in lane-half)
  #pragma unroll
  for (int g = 0; g < 8; ++g) {
    float mx = -1e30f;
    #pragma unroll
    for (int t = 0; t < 4; ++t) { sc[t][g] *= 0.25f; mx = fmaxf(mx, sc[t][g]); }
    mx = fmaxf(mx, __shfl_xor(mx, 1, 32));
    mx = fmaxf(mx, __shfl_xor(mx, 2, 32));
    mx = fmaxf(mx, __shfl_xor(mx, 4, 32));
    mx = fmaxf(mx, __shfl_xor(mx, 8, 32));
    float sm = 0.f;
    #pragma unroll
    for (int t = 0; t < 4; ++t) { float e = __expf(sc[t][g] - mx); sc[t][g] = e; sm += e; }
    sm += __shfl_xor(sm, 1, 32);
    sm += __shfl_xor(sm, 2, 32);
    sm += __shfl_xor(sm, 4, 32);
    sm += __shfl_xor(sm, 8, 32);
    float iv = 1.f / sm;
    #pragma unroll
    for (int t = 0; t < 4; ++t)
      lp[(g + 8 * hi) * 64 + nloc + 16 * t] = (_Float16)(sc[t][g] * iv);
  }
  __syncthreads();

  // O = P @ V  (K = 64 keys = two WMMA k-steps)
  v8f ao = V8F_ZERO;
  #pragma unroll
  for (int kb = 0; kb < 2; ++kb) {
    const _Float16* prow = lp + nloc * 64 + kb * 32;
    v8h pl = *(const v8h*)(prow + hi * 8);
    v8h ph = *(const v8h*)(prow + 16 + hi * 8);
    v16h ap;
    #pragma unroll
    for (int i = 0; i < 8; ++i) { ap[i] = pl[i]; ap[i + 8] = ph[i]; }
    v16h bv;
    #pragma unroll
    for (int i = 0; i < 16; ++i) {
      int key = kb * 32 + ((i < 8) ? (8 * hi + i) : (16 + 8 * hi + (i - 8)));
      bv[i] = qkv[(size_t)(b * 64 + key) * 384 + 256 + h * 16 + nloc];
    }
    ao = __builtin_amdgcn_wmma_f32_16x16x32_f16(false, ap, false, bv,
                                                (short)0, ao, false, false);
  }
  #pragma unroll
  for (int g = 0; g < 8; ++g) {
    int s = s0 + g + 8 * hi;
    att[(size_t)(b * 64 + s) * 128 + h * 16 + nloc] = (_Float16)ao[g];
  }
}

// mean over sequence (64) -> (16,128) f16
__global__ void k_mean(const _Float16* __restrict__ attp, _Float16* __restrict__ fmean) {
  int idx = blockIdx.x * 256 + threadIdx.x;
  if (idx >= 2048) return;
  int b = idx >> 7, e = idx & 127;
  float s = 0.f;
  for (int q = 0; q < 64; ++q) s += (float)attp[(size_t)(b * 64 + q) * 128 + e];
  fmean[idx] = (_Float16)(s * (1.f / 64.f));
}

// ---------------------------------------------------------------------------
extern "C" void kernel_launch(void* const* d_in, const int* in_sizes, int n_in,
                              void* d_out, int out_size, void* d_ws, size_t ws_size,
                              hipStream_t stream) {
  (void)in_sizes; (void)n_in; (void)out_size; (void)ws_size;
  // Input order: x first, then params leaves depth-first with sorted dict keys
  // (jax pytree order): attn{in_b,in_w,out_b,out_w}, bn1a{b,g,m,v}, bn1b{...},
  // bn2a, bn2b, c1a_b, c1a_w, c1b_b, c1b_w, c2a_b, c2a_w, c2b_b, c2b_w,
  // cls{bnf{b,g,m,v},fc1_b,fc1_w,fc2_b,fc2_w,fc3_b,fc3_w}, qparams.
  const float* x      = (const float*)d_in[0];
  const float* in_b   = (const float*)d_in[1];
  const float* in_w   = (const float*)d_in[2];
  const float* out_b  = (const float*)d_in[3];
  const float* out_w  = (const float*)d_in[4];
  const float* bn1a_b = (const float*)d_in[5],  *bn1a_g = (const float*)d_in[6];
  const float* bn1a_m = (const float*)d_in[7],  *bn1a_v = (const float*)d_in[8];
  const float* bn1b_b = (const float*)d_in[9],  *bn1b_g = (const float*)d_in[10];
  const float* bn1b_m = (const float*)d_in[11], *bn1b_v = (const float*)d_in[12];
  const float* bn2a_b = (const float*)d_in[13], *bn2a_g = (const float*)d_in[14];
  const float* bn2a_m = (const float*)d_in[15], *bn2a_v = (const float*)d_in[16];
  const float* bn2b_b = (const float*)d_in[17], *bn2b_g = (const float*)d_in[18];
  const float* bn2b_m = (const float*)d_in[19], *bn2b_v = (const float*)d_in[20];
  const float* c1a_b  = (const float*)d_in[21], *c1a_w  = (const float*)d_in[22];
  const float* c1b_b  = (const float*)d_in[23], *c1b_w  = (const float*)d_in[24];
  const float* c2a_b  = (const float*)d_in[25], *c2a_w  = (const float*)d_in[26];
  const float* c2b_b  = (const float*)d_in[27], *c2b_w  = (const float*)d_in[28];
  const float* bnf_b  = (const float*)d_in[29], *bnf_g  = (const float*)d_in[30];
  const float* bnf_m  = (const float*)d_in[31], *bnf_v  = (const float*)d_in[32];
  const float* fc1_b  = (const float*)d_in[33], *fc1_w  = (const float*)d_in[34];
  const float* fc2_b  = (const float*)d_in[35], *fc2_w  = (const float*)d_in[36];
  const float* fc3_b  = (const float*)d_in[37], *fc3_w  = (const float*)d_in[38];
  const float* qparams= (const float*)d_in[39];

  // -------- workspace bump allocator (256B aligned) --------
  size_t off = 0;
  auto alloc = [&](size_t bytes) -> char* {
    size_t o = (off + 255) & ~(size_t)255;
    off = o + bytes;
    return (char*)d_ws + o;
  };
  _Float16* A1    = (_Float16*)alloc((size_t)16*16*256*256*2);
  _Float16* A1b   = (_Float16*)alloc((size_t)16*16*256*256*2);
  _Float16* P1    = (_Float16*)alloc((size_t)16*16*128*128*2);
  _Float16* A2a   = (_Float16*)alloc((size_t)16*32*128*128*2);
  _Float16* A2b   = (_Float16*)alloc((size_t)16*32*128*128*2);
  _Float16* X2    = (_Float16*)alloc((size_t)16*32*64*64*2);
  float*    qct   = (float*)   alloc((size_t)16*8*4*32*32*4);
  _Float16* fbuf  = (_Float16*)alloc((size_t)16*8192*2);
  _Float16* qkv   = (_Float16*)alloc((size_t)1024*384*2);
  _Float16* att   = (_Float16*)alloc((size_t)1024*128*2);
  _Float16* attp  = (_Float16*)alloc((size_t)1024*128*2);
  _Float16* fmean = (_Float16*)alloc((size_t)2048*2);
  _Float16* h1    = (_Float16*)alloc((size_t)16*256*2);
  _Float16* h2    = (_Float16*)alloc((size_t)16*128*2);
  float*    Ur    = (float*)alloc(256*4);
  float*    Ui    = (float*)alloc(256*4);
  float *sc1a = (float*)alloc(16*4),  *sh1a = (float*)alloc(16*4);
  float *sc1b = (float*)alloc(16*4),  *sh1b = (float*)alloc(16*4);
  float *sc2a = (float*)alloc(32*4),  *sh2a = (float*)alloc(32*4);
  float *sc2b = (float*)alloc(32*4),  *sh2b = (float*)alloc(32*4);
  float *scf  = (float*)alloc(256*4), *shf  = (float*)alloc(256*4);
  _Float16* fg_c1a = (_Float16*)alloc((size_t)1*1*32*16*2);
  _Float16* fg_c1b = (_Float16*)alloc((size_t)1*5*32*16*2);
  _Float16* fg_c2a = (_Float16*)alloc((size_t)2*5*32*16*2);
  _Float16* fg_c2b = (_Float16*)alloc((size_t)2*9*32*16*2);
  _Float16* fg_inw = (_Float16*)alloc((size_t)24*4*32*16*2);
  _Float16* fg_outw= (_Float16*)alloc((size_t)8*4*32*16*2);
  _Float16* fg_fc1 = (_Float16*)alloc((size_t)16*4*32*16*2);
  _Float16* fg_fc2 = (_Float16*)alloc((size_t)8*8*32*16*2);
  _Float16* fg_fc3 = (_Float16*)alloc((size_t)1*4*32*16*2);

  // -------- parameter preprocessing --------
  k_bnprep<<<1, 256, 0, stream>>>(bn1a_g, bn1a_b, bn1a_m, bn1a_v, c1a_b, sc1a, sh1a, 16);
  k_bnprep<<<1, 256, 0, stream>>>(bn1b_g, bn1b_b, bn1b_m, bn1b_v, c1b_b, sc1b, sh1b, 16);
  k_bnprep<<<1, 256, 0, stream>>>(bn2a_g, bn2a_b, bn2a_m, bn2a_v, c2a_b, sc2a, sh2a, 32);
  k_bnprep<<<1, 256, 0, stream>>>(bn2b_g, bn2b_b, bn2b_m, bn2b_v, c2b_b, sc2b, sh2b, 32);
  k_bnprep<<<1, 256, 0, stream>>>(bnf_g, bnf_b, bnf_m, bnf_v, nullptr, scf, shf, 256);
  k_prep_wfrag<<<1,  256, 0, stream>>>(c1a_w, fg_c1a, 16, 9);
  k_prep_wfrag<<<1,  256, 0, stream>>>(c1b_w, fg_c1b, 16, 144);
  k_prep_wfrag<<<2,  256, 0, stream>>>(c2a_w, fg_c2a, 32, 144);
  k_prep_wfrag<<<3,  256, 0, stream>>>(c2b_w, fg_c2b, 32, 288);
  k_prep_wfrag<<<12, 256, 0, stream>>>(in_w,  fg_inw, 384, 128);
  k_prep_wfrag<<<4,  256, 0, stream>>>(out_w, fg_outw,128, 128);
  k_prep_wfrag<<<8,  256, 0, stream>>>(fc1_w, fg_fc1, 256, 128);
  k_prep_wfrag<<<8,  256, 0, stream>>>(fc2_w, fg_fc2, 128, 256);
  k_prep_wfrag<<<1,  256, 0, stream>>>(fc3_w, fg_fc3, 10, 128);
  k_qunitary<<<1, 32, 0, stream>>>(qparams, Ur, Ui);

  // -------- conv stack --------
  k_conv3x3_wmma<1, 16, true ><<<16*256*16, 32, 0, stream>>>(x,   fg_c1a, sc1a, sh1a, A1,  16, 256, 256);
  k_conv3x3_wmma<16,16, false><<<16*256*16, 32, 0, stream>>>(A1,  fg_c1b, sc1b, sh1b, A1b, 16, 256, 256);
  k_pool2<<<(16*16*128*128 + 255)/256, 256, 0, stream>>>(A1b, P1, 16*16, 128, 128);
  k_conv3x3_wmma<16,32, false><<<16*128*8,  32, 0, stream>>>(P1,  fg_c2a, sc2a, sh2a, A2a, 16, 128, 128);
  k_conv3x3_wmma<32,32, false><<<16*128*8,  32, 0, stream>>>(A2a, fg_c2b, sc2b, sh2b, A2b, 16, 128, 128);
  k_pool2<<<(16*32*64*64 + 255)/256, 256, 0, stream>>>(A2b, X2, 16*32, 64, 64);

  // -------- quantum branch --------
  k_qstage1<<<(16*8*32*32 + 255)/256, 256, 0, stream>>>(X2, Ur, Ui, qct);
  k_qstage2<<<(16*8*4*16*16 + 255)/256, 256, 0, stream>>>(qct, fbuf);

  // -------- MHA --------
  k_gemm16<0><<<dim3(64, 24), 32, 0, stream>>>(fbuf, 128, fg_inw, in_b,
                                               nullptr, nullptr, qkv, nullptr, 384, 384);
  k_attn<<<512, 32, 0, stream>>>(qkv, att);
  k_gemm16<0><<<dim3(64, 8), 32, 0, stream>>>(att, 128, fg_outw, out_b,
                                              nullptr, nullptr, attp, nullptr, 128, 128);
  k_mean<<<8, 256, 0, stream>>>(attp, fmean);

  // -------- classifier --------
  k_gemm16<1><<<dim3(1, 16), 32, 0, stream>>>(fmean, 128, fg_fc1, fc1_b,
                                              scf, shf, h1, nullptr, 256, 256);
  k_gemm16<1><<<dim3(1, 8), 32, 0, stream>>>(h1, 256, fg_fc2, fc2_b,
                                             nullptr, nullptr, h2, nullptr, 128, 128);
  k_gemm16<0><<<dim3(1, 1), 32, 0, stream>>>(h2, 128, fg_fc3, fc3_b,
                                             nullptr, nullptr, nullptr, (float*)d_out, 10, 10);
}
